// GRUNet_16131897164359
// MI455X (gfx1250) — compile-verified, matrix operand order checked by reading, not run on hardware
//
#include <hip/hip_runtime.h>
#include <hip/hip_bf16.h>

typedef __bf16 bf16;
typedef __attribute__((ext_vector_type(16))) __bf16 v16bf;
typedef __attribute__((ext_vector_type(8)))  __bf16 bf16x8;
typedef __attribute__((ext_vector_type(8)))  float   v8f;

#define B_  256
#define T_  104
#define I_  512
#define H_  512
#define O_  1024
#define G3  1536            // 3*H
#define HS  53760           // 105*H : row stride of concatenated hidden states
#define XRS (T_ * I_)       // 53248 : row stride of x viewed as (B, T*I)

// ---------------------------------------------------------------------------
// Fragment helpers for V_WMMA_F32_16X16X32_BF16 (wave32).
// 16-bit A-matrix 16x32 layout (ISA 7.12.2): lane L<16 -> row L, K 0..7 in
// halves 0..7 and K 16..23 in halves 8..15; lane L+16 -> row L, K 8..15 /
// K 24..31.  B (KxN) loaded as rows of W (N x K row-major) == B^T, same
// striping.  All tile bases/strides below are 16-byte aligned.
// ---------------------------------------------------------------------------
__device__ __forceinline__ v16bf load_frag16x32(const bf16* tile, size_t rstride, int lane) {
  int r  = lane & 15;
  int kb = (lane >> 4) << 3;              // 0 or 8
  const bf16* p = tile + (size_t)r * rstride + kb;
  bf16x8 lo = *(const bf16x8*)(p);
  bf16x8 hi = *(const bf16x8*)(p + 16);
  v16bf f;
#pragma unroll
  for (int i = 0; i < 8; ++i) { f[i] = lo[i]; f[i + 8] = hi[i]; }
  return f;
}

__device__ __forceinline__ v8f wmma_bf16(v16bf a, v16bf b, v8f c) {
  return __builtin_amdgcn_wmma_f32_16x16x32_bf16(false, a, false, b, (short)0, c, false, false);
}

// C/D 16x16 f32 layout: VGPR r -> M = r + 8*(lane>=16), N = lane&15.
__device__ __forceinline__ void store_acc_bias(float* tile, size_t rstride,
                                               const v8f& acc, int lane, float bias) {
  int nl = lane & 15;
  int mh = (lane >> 4) * 8;
#pragma unroll
  for (int rv = 0; rv < 8; ++rv)
    tile[(size_t)(mh + rv) * rstride + nl] = acc[rv] + bias;
}

__device__ __forceinline__ float fast_sigmoid(float x) { return 1.0f / (1.0f + __expf(-x)); }
__device__ __forceinline__ float fast_tanh(float x) {
  float e = __expf(-2.0f * x);
  return (1.0f - e) / (1.0f + e);
}

// ---------------------------------------------------------------------------
// Kernel 0: fp32 -> bf16 conversions + zero h0 block of hidden buffer.
// ---------------------------------------------------------------------------
__global__ __launch_bounds__(256) void prep_kernel(
    const float* __restrict__ x, const float* __restrict__ w_ih,
    const float* __restrict__ w_hh, const float* __restrict__ fc_w,
    bf16* __restrict__ x_bf, bf16* __restrict__ wih_bf, bf16* __restrict__ whh_bf,
    bf16* __restrict__ fcw_bf, bf16* __restrict__ hs_bf) {
  size_t tid    = (size_t)blockIdx.x * blockDim.x + threadIdx.x;
  size_t stride = (size_t)gridDim.x * blockDim.x;
  const size_t n_wih = (size_t)G3 * I_;
  const size_t n_whh = (size_t)G3 * H_;
  const size_t n_x   = (size_t)B_ * T_ * I_;
  const size_t n_fcw = (size_t)O_ * HS;
  for (size_t i = tid; i < n_wih; i += stride) wih_bf[i] = (bf16)w_ih[i];
  for (size_t i = tid; i < n_whh; i += stride) whh_bf[i] = (bf16)w_hh[i];
  for (size_t i = tid; i < n_x;   i += stride) x_bf[i]   = (bf16)x[i];
  for (size_t i = tid; i < n_fcw; i += stride) fcw_bf[i] = (bf16)fc_w[i];
  for (size_t i = tid; i < (size_t)B_ * H_; i += stride) {
    size_t b = i >> 9, c = i & 511;
    hs_bf[b * HS + c] = (bf16)0.0f;
  }
}

// ---------------------------------------------------------------------------
// Kernel 1: x_proj[t][b][g] = sum_i x[b,t,i]*w_ih[g,i] + b_ih[g]   (f32 out)
// grid (T, B/64, 3H/256), 8 waves: 2(M)x4(N), wave tile 32x64.
// w_ih (1.5 MB) is L2-resident; direct global fragment loads pipeline fine.
// ---------------------------------------------------------------------------
__global__ __launch_bounds__(256) void xproj_kernel(
    const bf16* __restrict__ x_bf, const bf16* __restrict__ wih_bf,
    const float* __restrict__ b_ih, float* __restrict__ x_proj) {
  const int t    = blockIdx.x;
  const int mb   = blockIdx.y * 64;
  const int nb   = blockIdx.z * 256;
  const int lane = threadIdx.x & 31;
  const int wave = threadIdx.x >> 5;
  const int m0   = mb + (wave >> 2) * 32;
  const int n0   = nb + (wave & 3) * 64;

  v8f acc[2][4];
#pragma unroll
  for (int i = 0; i < 2; ++i)
#pragma unroll
    for (int j = 0; j < 4; ++j) acc[i][j] = (v8f){};

  for (int k = 0; k < I_; k += 32) {
    v16bf a0 = load_frag16x32(x_bf + (size_t)m0 * XRS + (size_t)t * I_ + k, XRS, lane);
    v16bf a1 = load_frag16x32(x_bf + (size_t)(m0 + 16) * XRS + (size_t)t * I_ + k, XRS, lane);
#pragma unroll
    for (int j = 0; j < 4; ++j) {
      v16bf bq = load_frag16x32(wih_bf + (size_t)(n0 + 16 * j) * I_ + k, I_, lane);
      acc[0][j] = wmma_bf16(a0, bq, acc[0][j]);
      acc[1][j] = wmma_bf16(a1, bq, acc[1][j]);
    }
  }

#pragma unroll
  for (int i = 0; i < 2; ++i)
#pragma unroll
    for (int j = 0; j < 4; ++j) {
      int   c0   = n0 + 16 * j;
      float bias = b_ih[c0 + (lane & 15)];
      float* tile = x_proj + ((size_t)t * B_ + (m0 + 16 * i)) * G3 + c0;
      store_acc_bias(tile, G3, acc[i][j], lane, bias);
    }
}

// ---------------------------------------------------------------------------
// Kernel 2: one GRU step. Each wave: 16(batch) x 32(hidden) output, all 3
// gates accumulated (6 WMMA / K-step). Reads h_t from hs_bf col t*H, writes
// h_{t+1} bf16 to col (t+1)*H.  grid (B/32, H/128), 8 waves 2(M)x4(N).
// ---------------------------------------------------------------------------
__global__ __launch_bounds__(256) void gru_step_kernel(
    int t, const bf16* __restrict__ whh_bf, const float* __restrict__ b_hh,
    const float* __restrict__ x_proj, bf16* __restrict__ hs_bf) {
  const int mb   = blockIdx.x * 32;
  const int hb0  = blockIdx.y * 128;
  const int lane = threadIdx.x & 31;
  const int wave = threadIdx.x >> 5;
  const int m0   = mb + (wave >> 2) * 16;
  const int hb   = hb0 + (wave & 3) * 32;

  v8f acc[2][3];
#pragma unroll
  for (int j = 0; j < 2; ++j)
#pragma unroll
    for (int g = 0; g < 3; ++g) acc[j][g] = (v8f){};

  const size_t hcol = (size_t)t * H_;
  for (int k = 0; k < H_; k += 32) {
    v16bf a = load_frag16x32(hs_bf + (size_t)m0 * HS + hcol + k, HS, lane);
#pragma unroll
    for (int j = 0; j < 2; ++j) {
#pragma unroll
      for (int g = 0; g < 3; ++g) {
        v16bf bq = load_frag16x32(whh_bf + (size_t)(g * H_ + hb + 16 * j) * H_ + k, H_, lane);
        acc[j][g] = wmma_bf16(a, bq, acc[j][g]);
      }
    }
  }

  const int nl = lane & 15;
  const int mh = (lane >> 4) * 8;
#pragma unroll
  for (int j = 0; j < 2; ++j) {
    const int nn  = hb + 16 * j + nl;
    const float bhr = b_hh[nn];
    const float bhz = b_hh[H_ + nn];
    const float bhn = b_hh[2 * H_ + nn];
#pragma unroll
    for (int rv = 0; rv < 8; ++rv) {
      const int m = m0 + mh + rv;
      const float* xp = x_proj + ((size_t)t * B_ + m) * G3;
      float xr = xp[nn], xz = xp[H_ + nn], xn = xp[2 * H_ + nn];
      float r  = fast_sigmoid(xr + acc[j][0][rv] + bhr);
      float z  = fast_sigmoid(xz + acc[j][1][rv] + bhz);
      float nv = fast_tanh(xn + r * (acc[j][2][rv] + bhn));
      float hp = (float)hs_bf[(size_t)m * HS + hcol + nn];
      float hn = (1.0f - z) * nv + z * hp;
      hs_bf[(size_t)m * HS + hcol + H_ + nn] = (bf16)hn;
    }
  }
}

// ---------------------------------------------------------------------------
// Kernel 3: FC partials with split-K=8 using ASYNC global->LDS staging.
// grid (B/64, O/256, 8), 8 waves 2x4, wave tile 32x64, wg tile 64x256.
// Tiles are staged once per WG (double-buffered) with
// GLOBAL_LOAD_ASYNC_TO_LDS_B128 (ASYNCcnt), fragments re-read from LDS.
// LDS rows padded to 40 bf16 (80 B): lane r hits banks 20r mod 64 -> all 16
// lanes on disjoint 4-bank groups, conflict-free ds_load_b128.
// ---------------------------------------------------------------------------
#define KC_LEN (HS / 8)     // 6720
#define KC_NIT (KC_LEN / 32) // 210
#define SP 40               // padded LDS row stride (elements)

__device__ __forceinline__ void async_ld16(const bf16* gp, bf16* lp) {
  uint32_t lds = (uint32_t)(uintptr_t)lp;            // low 32 bits = LDS offset
  uint64_t ga  = (uint64_t)(uintptr_t)gp;
  asm volatile("global_load_async_to_lds_b128 %0, %1, off"
               :: "v"(lds), "v"(ga) : "memory");
}

// Stage A (64x32) + B (256x32) K-tiles: 1280 x 16B transfers, 5 per thread.
__device__ __forceinline__ void stage_tiles(const bf16* __restrict__ hsrow,
                                            const bf16* __restrict__ fcrow,
                                            bf16* smA, bf16* smB, int k, int tid) {
  {
    int row = tid >> 2, seg = tid & 3;               // A: 64 rows * 4 segs
    async_ld16(hsrow + (size_t)row * HS + k + seg * 8, smA + row * SP + seg * 8);
  }
#pragma unroll
  for (int it = 0; it < 4; ++it) {                   // B: 256 rows * 4 segs
    int s = tid + 256 * it;
    int row = s >> 2, seg = s & 3;
    async_ld16(fcrow + (size_t)row * HS + k + seg * 8, smB + row * SP + seg * 8);
  }
}

__global__ __launch_bounds__(256) void fc_kernel(
    const bf16* __restrict__ hs_bf, const bf16* __restrict__ fcw_bf,
    float* __restrict__ part) {
  __shared__ bf16 smA[2][64 * SP];    // 2 x  5.0 KB
  __shared__ bf16 smB[2][256 * SP];   // 2 x 20.0 KB

  const int tid  = threadIdx.x;
  const int mb   = blockIdx.x * 64;
  const int nb   = blockIdx.y * 256;
  const int kc   = blockIdx.z;
  const int lane = tid & 31;
  const int wave = tid >> 5;
  const int wm   = (wave >> 2) * 32;    // wave M offset inside wg tile
  const int wn   = (wave & 3) * 64;     // wave N offset inside wg tile

  const bf16* hsrow = hs_bf  + (size_t)mb * HS;
  const bf16* fcrow = fcw_bf + (size_t)nb * HS;

  v8f acc[2][4];
#pragma unroll
  for (int i = 0; i < 2; ++i)
#pragma unroll
    for (int j = 0; j < 4; ++j) acc[i][j] = (v8f){};

  const int k0 = kc * KC_LEN;
  stage_tiles(hsrow, fcrow, smA[0], smB[0], k0, tid);
  asm volatile("s_wait_asynccnt 0x0" ::: "memory");
  __syncthreads();

  for (int kt = 0; kt < KC_NIT; ++kt) {
    const int cur = kt & 1;
    if (kt + 1 < KC_NIT)
      stage_tiles(hsrow, fcrow, smA[cur ^ 1], smB[cur ^ 1],
                  k0 + (kt + 1) * 32, tid);

    v16bf a0 = load_frag16x32(&smA[cur][(wm + 0)  * SP], SP, lane);
    v16bf a1 = load_frag16x32(&smA[cur][(wm + 16) * SP], SP, lane);
#pragma unroll
    for (int j = 0; j < 4; ++j) {
      v16bf bq = load_frag16x32(&smB[cur][(wn + 16 * j) * SP], SP, lane);
      acc[0][j] = wmma_bf16(a0, bq, acc[0][j]);
      acc[1][j] = wmma_bf16(a1, bq, acc[1][j]);
    }

    asm volatile("s_wait_asynccnt 0x0" ::: "memory");
    __syncthreads();   // DScnt-fenced barrier: buffers safe to swap
  }

#pragma unroll
  for (int i = 0; i < 2; ++i)
#pragma unroll
    for (int j = 0; j < 4; ++j) {
      float* tile = part + ((size_t)kc * B_ + (mb + wm + 16 * i)) * O_
                         + (nb + wn + 16 * j);
      store_acc_bias(tile, O_, acc[i][j], lane, 0.0f);
    }
}

// ---------------------------------------------------------------------------
// Kernel 4: out[b][o] = fc_b[o] + sum_kc part[kc][b][o]
// ---------------------------------------------------------------------------
__global__ __launch_bounds__(256) void reduce_kernel(
    const float* __restrict__ part, const float* __restrict__ fc_b,
    float* __restrict__ out) {
  size_t idx = (size_t)blockIdx.x * blockDim.x + threadIdx.x;
  if (idx >= (size_t)B_ * O_) return;
  int o = (int)(idx & (O_ - 1));
  float s = fc_b[o];
#pragma unroll
  for (int kc = 0; kc < 8; ++kc) s += part[(size_t)kc * B_ * O_ + idx];
  out[idx] = s;
}

// ---------------------------------------------------------------------------
extern "C" void kernel_launch(void* const* d_in, const int* in_sizes, int n_in,
                              void* d_out, int out_size, void* d_ws, size_t ws_size,
                              hipStream_t stream) {
  const float* x    = (const float*)d_in[0];
  const float* w_ih = (const float*)d_in[1];
  const float* w_hh = (const float*)d_in[2];
  const float* b_ih = (const float*)d_in[3];
  const float* b_hh = (const float*)d_in[4];
  const float* fc_w = (const float*)d_in[5];
  const float* fc_b = (const float*)d_in[6];
  float* out = (float*)d_out;

  // workspace layout (bytes), ~340 MB total
  char* ws = (char*)d_ws;
  bf16*  wih_bf = (bf16*)(ws + 0);                         //  1.5 MB
  bf16*  whh_bf = (bf16*)(ws + 1572864);                   //  1.5 MB
  bf16*  x_bf   = (bf16*)(ws + 3145728);                   // 27.3 MB
  bf16*  fcw_bf = (bf16*)(ws + 30408704);                  // 110.1 MB
  bf16*  hs_bf  = (bf16*)(ws + 140509184);                 // 27.5 MB
  float* x_proj = (float*)(ws + 168034304);                // 163.6 MB
  float* part   = (float*)(ws + 331612160);                //  8.4 MB

  prep_kernel<<<2048, 256, 0, stream>>>(x, w_ih, w_hh, fc_w,
                                        x_bf, wih_bf, whh_bf, fcw_bf, hs_bf);

  xproj_kernel<<<dim3(T_, B_ / 64, G3 / 256), 256, 0, stream>>>(
      x_bf, wih_bf, b_ih, x_proj);

  for (int t = 0; t < T_; ++t)
    gru_step_kernel<<<dim3(B_ / 32, H_ / 128), 256, 0, stream>>>(
        t, whh_bf, b_hh, x_proj, hs_bf);

  fc_kernel<<<dim3(B_ / 64, O_ / 256, 8), 256, 0, stream>>>(hs_bf, fcw_bf, part);

  reduce_kernel<<<(B_ * O_ + 255) / 256, 256, 0, stream>>>(part, fc_b, out);
}
